// CharLSTM_67808943669333
// MI455X (gfx1250) — matrix operand order repaired and checked by reading, not compiled, not run on hardware
//
#include <hip/hip_runtime.h>

// ---------------------------------------------------------------------------
// Tree-LSTM fused cell for MI455X (gfx1250), wave32 + WMMA bf16.
//   gates (i,o,c) via v_wmma_f32_16x16x32_bf16, f32 accumulate
//   f-gate and c_pre are dead code in the reference -> skipped entirely
//   MT=32 rows/WG: each wave owns 2 row-tiles x 6 col-tiles so every B
//   fragment loaded from L2 feeds 2 WMMAs (halves L2 fragment traffic).
// ---------------------------------------------------------------------------

typedef __attribute__((ext_vector_type(16))) __bf16 v16bf;
typedef __attribute__((ext_vector_type(8)))  __bf16 v8bf;
typedef __attribute__((ext_vector_type(8)))  float  v8f;
typedef __attribute__((ext_vector_type(4)))  float  v4f;

#define B_ROWS   32768
#define E_DIM    128
#define H_DIM    256
#define KCH      8
#define KD       384            // E + H
#define NCOLS    768            // 3 gates * 256
#define KT_COUNT 12             // KD / 32
#define XSTRIDE  392            // 384 + 8 pad (bf16 elems) -> 2-phase-min ds_load_b128
#define GSTRIDE  772            // 768 + 4 pad (f32)
#define MT       32             // batch rows per workgroup (2 row-tiles)
#define LDS_BYTES (MT * GSTRIDE * 4)   // 98,816 B; xh (25,088 B) aliases the front

__device__ __forceinline__ __bf16 f2bf(float f) {
  union { float f; unsigned u; } v; v.f = f;
  unsigned r = v.u + 0x7FFFu + ((v.u >> 16) & 1u);   // round-to-nearest-even
  union { unsigned short u; __bf16 b; } c; c.u = (unsigned short)(r >> 16);
  return c.b;
}

__device__ __forceinline__ float sigmoid_f(float x) {
  return 1.0f / (1.0f + __expf(-x));
}
__device__ __forceinline__ float tanh_f(float x) {
  return 1.0f - 2.0f / (__expf(2.0f * x) + 1.0f);
}

// ---------------------------------------------------------------------------
// Pre-swizzle [Wi;Wo;Wc] into bf16 WMMA B-fragments:
//   frags[ct][kt][lane][e]  (ct: 16-col tile over 768 cols, kt: 32-K tile)
//   lane<16 : N = lane,    K = kt*32 + e        (e = 0..15)
//   lane>=16: N = lane-16, K = kt*32 + 16 + e
// Total 48*12*32*16 = 294,912 bf16 = 576 KB in d_ws.
// ---------------------------------------------------------------------------
__global__ __launch_bounds__(256) void prepack_weights(
    const float* __restrict__ Wi, const float* __restrict__ Wo,
    const float* __restrict__ Wc, __bf16* __restrict__ frags) {
  int idx  = blockIdx.x * 256 + threadIdx.x;      // exact cover, no bounds check
  int e    = idx & 15;
  int lane = (idx >> 4) & 31;
  int t    = idx >> 9;
  int kt   = t % KT_COUNT;
  int ct   = t / KT_COUNT;                        // 0..47
  int g    = ct >> 4;                             // 0=i, 1=o, 2=c
  int n    = ((ct & 15) << 4) + (lane & 15);      // 0..255 within gate
  int k    = kt * 32 + ((lane >> 4) << 4) + e;    // 0..383
  const float* W = (g == 0) ? Wi : ((g == 1) ? Wo : Wc);
  frags[idx] = f2bf(W[n * KD + k]);
}

// ---------------------------------------------------------------------------
// Fused: embed gather + concat -> bf16 LDS -> 3-gate WMMA GEMM -> softmax over
// [i ; 8 children] -> c_cur, h_cur.  One workgroup = 32 batch rows, 8 waves,
// each wave owns 2 row-tiles x 6 col-tiles (48 col-tiles total).
// ---------------------------------------------------------------------------
__global__ __launch_bounds__(256) void fused_treelstm(
    const int*   __restrict__ x,
    const float* __restrict__ h_pre,
    const float* __restrict__ child_cells,
    const float* __restrict__ child_weights,
    const float* __restrict__ embed,
    const float* __restrict__ bi,
    const float* __restrict__ bo,
    const float* __restrict__ bc,
    const __bf16* __restrict__ frags,
    float* __restrict__ out) {
  extern __shared__ __align__(16) char smem[];
  __bf16* xh    = (__bf16*)smem;      // phase 1-2: [MT][XSTRIDE] bf16 (25 KB)
  float*  gates = (float*)smem;       // phase 3-4: [MT][GSTRIDE] f32 (96.5 KB)

  const int t = threadIdx.x;

  // ---- Phase 1: build xh tile (bf16) in LDS: [emb(128) | h_pre(256)] ------
  {
    const int b_local = t >> 3;              // 32 rows, 8 threads per row
    const int bg      = blockIdx.x * MT + b_local;
    const int sub     = t & 7;
    if (sub < 4) {
      int vidx = x[bg];
      const float* src = embed + (size_t)vidx * E_DIM + sub * 32;
      __bf16* dst = xh + b_local * XSTRIDE + sub * 32;
#pragma unroll
      for (int c = 0; c < 32; c += 4) {
        v4f v = *(const v4f*)(src + c);
        dst[c + 0] = f2bf(v.x); dst[c + 1] = f2bf(v.y);
        dst[c + 2] = f2bf(v.z); dst[c + 3] = f2bf(v.w);
      }
    } else {
      const float* src = h_pre + (size_t)bg * H_DIM + (sub - 4) * 64;
      __bf16* dst = xh + b_local * XSTRIDE + E_DIM + (sub - 4) * 64;
#pragma unroll
      for (int c = 0; c < 64; c += 4) {
        v4f v = *(const v4f*)(src + c);
        dst[c + 0] = f2bf(v.x); dst[c + 1] = f2bf(v.y);
        dst[c + 2] = f2bf(v.z); dst[c + 3] = f2bf(v.w);
      }
    }
  }
  __syncthreads();

  // ---- Phase 2: WMMA GEMM, C[32 x 768] over K=384 -------------------------
  const int lane = t & 31;
  const int w    = t >> 5;                         // wave id 0..7
  v8f acc0[6], acc1[6];
#pragma unroll
  for (int j = 0; j < 6; ++j) { acc0[j] = (v8f)0.0f; acc1[j] = (v8f)0.0f; }

  const __bf16* rowp0 = xh + (lane & 15) * XSTRIDE;   // row-tile 0: M = lane%16
  const __bf16* rowp1 = rowp0 + 16 * XSTRIDE;         // row-tile 1: M = 16 + lane%16
  const int     khalf = (lane >> 4) << 4;             // 0 (lanes 0-15) or 16
  const __bf16* fbase = frags + (size_t)(w * 6) * KT_COUNT * 512 + lane * 16;

  for (int kt = 0; kt < KT_COUNT; ++kt) {
    // A fragment in contiguous K order (matches prepacked B):
    //   lane<16 : e0..15 = K0..15
    //   lane>=16: e0..15 = K16..31
    const int k0 = kt * 32 + khalf;
    v8bf a0lo = *(const v8bf*)(rowp0 + k0);
    v8bf a0hi = *(const v8bf*)(rowp0 + k0 + 8);
    v8bf a1lo = *(const v8bf*)(rowp1 + k0);
    v8bf a1hi = *(const v8bf*)(rowp1 + k0 + 8);
    v16bf a0, a1;
#pragma unroll
    for (int e = 0; e < 8; ++e) {
      a0[e] = a0lo[e]; a0[e + 8] = a0hi[e];
      a1[e] = a1lo[e]; a1[e + 8] = a1hi[e];
    }

    // Load all 6 B fragments into distinct regs first so the scheduler can
    // clause the 12 global_load_b128 and overlap them with the WMMAs.
    v16bf bfr[6];
#pragma unroll
    for (int j = 0; j < 6; ++j)
      bfr[j] = *(const v16bf*)(fbase + (size_t)(j * KT_COUNT + kt) * 512);

#pragma unroll
    for (int j = 0; j < 6; ++j) {
      acc0[j] = __builtin_amdgcn_wmma_f32_16x16x32_bf16(
          false, a0, false, bfr[j], (short)0, acc0[j], false, false);
      acc1[j] = __builtin_amdgcn_wmma_f32_16x16x32_bf16(
          false, a1, false, bfr[j], (short)0, acc1[j], false, false);
    }
  }

  // xh region is about to be overwritten by gates (aliased LDS)
  __syncthreads();

  // ---- Phase 3: spill gate tiles to LDS (C layout: lanes16-31 -> M=8+r) ---
  {
    const int N     = lane & 15;
    const int mbase = (lane >> 4) << 3;
#pragma unroll
    for (int j = 0; j < 6; ++j) {
      const int col = (w * 6 + j) * 16 + N;
#pragma unroll
      for (int r = 0; r < 8; ++r) {
        gates[(mbase + r) * GSTRIDE + col]        = acc0[j][r];
        gates[(16 + mbase + r) * GSTRIDE + col]   = acc1[j][r];
      }
    }
  }
  __syncthreads();

  // ---- Phase 4: softmax over [i ; 8 children] + activations ---------------
  union F4 { v4f v; float f[4]; };
  const size_t BH      = (size_t)B_ROWS * H_DIM;
  const int    b_local = t >> 3;                 // 32 rows, 8 threads per row
  const int    bg      = blockIdx.x * MT + b_local;
  const float* grow    = gates + b_local * GSTRIDE;
  const int    h0      = (t & 7) * 32;           // 32 h-cols per thread

#pragma unroll
  for (int chunk = 0; chunk < 8; ++chunk) {
    const int    h    = h0 + chunk * 4;
    const size_t base = (size_t)bg * H_DIM + h;

    F4 gi, go, gc, vbi, vbo, vbc;
    gi.v  = *(const v4f*)(grow + h);
    go.v  = *(const v4f*)(grow + H_DIM + h);
    gc.v  = *(const v4f*)(grow + 2 * H_DIM + h);
    vbi.v = *(const v4f*)(bi + h);
    vbo.v = *(const v4f*)(bo + h);
    vbc.v = *(const v4f*)(bc + h);

    F4 cw[KCH], cc[KCH];
#pragma unroll
    for (int k = 0; k < KCH; ++k) {   // streamed once -> non-temporal
      cw[k].v = __builtin_nontemporal_load((const v4f*)(child_weights + k * BH + base));
      cc[k].v = __builtin_nontemporal_load((const v4f*)(child_cells   + k * BH + base));
    }

    F4 hout, cout;
#pragma unroll
    for (int c = 0; c < 4; ++c) {
      float iv = sigmoid_f(gi.f[c] + vbi.f[c]);
      float ov = sigmoid_f(go.f[c] + vbo.f[c]);
      float cv = tanh_f(gc.f[c] + vbc.f[c]);
      float m = iv;
#pragma unroll
      for (int k = 0; k < KCH; ++k) m = fmaxf(m, cw[k].f[c]);
      float s   = __expf(iv - m);
      float num = s * cv;
#pragma unroll
      for (int k = 0; k < KCH; ++k) {
        float ek = __expf(cw[k].f[c] - m);
        s   += ek;
        num += ek * cc[k].f[c];
      }
      float ccur = num / s;
      hout.f[c] = ov * tanh_f(ccur);
      cout.f[c] = ccur;
    }
    __builtin_nontemporal_store(hout.v, (v4f*)(out + base));
    __builtin_nontemporal_store(cout.v, (v4f*)(out + BH + base));
  }
}

// ---------------------------------------------------------------------------
extern "C" void kernel_launch(void* const* d_in, const int* in_sizes, int n_in,
                              void* d_out, int out_size, void* d_ws, size_t ws_size,
                              hipStream_t stream) {
  const int*   x             = (const int*)  d_in[0];
  const float* h_pre         = (const float*)d_in[1];
  // d_in[2] = c_pre : unused by the reference outputs
  const float* child_cells   = (const float*)d_in[3];
  const float* child_weights = (const float*)d_in[4];
  const float* embed         = (const float*)d_in[5];
  // d_in[6] = Wf, d_in[7] = bf : dead code in the reference
  const float* Wi = (const float*)d_in[8];
  const float* bi = (const float*)d_in[9];
  const float* Wo = (const float*)d_in[10];
  const float* bo = (const float*)d_in[11];
  const float* Wc = (const float*)d_in[12];
  const float* bc = (const float*)d_in[13];

  __bf16* frags = (__bf16*)d_ws;         // 576 KB bf16 fragment buffer
  float*  out   = (float*)d_out;

  // 48*12*32*16 / 256 = 1152 blocks
  prepack_weights<<<1152, 256, 0, stream>>>(Wi, Wo, Wc, frags);
  fused_treelstm<<<B_ROWS / MT, 256, LDS_BYTES, stream>>>(
      x, h_pre, child_cells, child_weights, embed, bi, bo, bc, frags, out);
}
